// DenseEncoderLayer_76527727280618
// MI455X (gfx1250) — matrix operand order repaired and checked by reading, not compiled, not run on hardware
//
#include <hip/hip_runtime.h>
#include <hip/hip_bf16.h>
#include <math.h>

typedef __attribute__((ext_vector_type(16))) _Float16 v16h;
typedef __attribute__((ext_vector_type(8)))  float    v8f;

#define SEQ   2048
#define DIM   512
#define DH    64
#define NH    8
#define NE    8
#define FF    2048

#define BM 128
#define BK 32

// ---------------------------------------------------------------- utilities
__global__ __launch_bounds__(256) void zero_f32_kernel(float* __restrict__ p, int n) {
    int i = blockIdx.x * 256 + threadIdx.x;
    if (i < n) p[i] = 0.0f;
}

// tanh-approx GELU in sigmoid form: 0.5x(1+tanh(u)) == x*sigmoid(2u)
__device__ __forceinline__ float gelu_tanh(float x) {
    float t = x * (1.5957691216057308f + 0.07135481283146122f * x * x);  // 2u
    return x / (1.0f + __expf(-t));
}

// low 32 bits of a generic pointer into LDS == LDS byte offset
__device__ __forceinline__ unsigned lds_off(const void* p) {
    return (unsigned)(uintptr_t)p;
}

// CDNA5 async DMA: global -> LDS, 16 bytes per lane, tracked by ASYNCcnt
__device__ __forceinline__ void async_copy_b128(unsigned lds, const _Float16* g) {
    asm volatile("global_load_async_to_lds_b128 %0, %1, off"
                 :: "v"(lds), "v"(g) : "memory");
}

// ---------------------------------------------------------------- transpose + convert
// in: f32 [R][C] (+ batch*R*C), out: f16 [C][R] (+ batch*R*C). 32x32 LDS tile.
__global__ __launch_bounds__(256) void f32_to_f16t_kernel(const float* __restrict__ in,
                                                          _Float16* __restrict__ out,
                                                          int R, int C) {
    __shared__ float t[32][33];
    const int tx = threadIdx.x & 31, ty = threadIdx.x >> 5;   // 32 x 8
    const int r0 = blockIdx.y * 32, c0 = blockIdx.x * 32;
    const size_t zo = (size_t)blockIdx.z * R * C;
    const float* pin = in + zo;
    _Float16* pout = out + zo;
    #pragma unroll
    for (int j = 0; j < 4; ++j)
        t[ty + j * 8][tx] = pin[(size_t)(r0 + ty + j * 8) * C + c0 + tx];
    __syncthreads();
    #pragma unroll
    for (int j = 0; j < 4; ++j)
        pout[(size_t)(c0 + ty + j * 8) * R + r0 + tx] = (_Float16)t[tx][ty + j * 8];
}

// ---------------------------------------------------------------- layernorm -> f16
__global__ __launch_bounds__(256) void layernorm_f16_kernel(const float* __restrict__ x,
                                                            const float* __restrict__ g,
                                                            const float* __restrict__ b,
                                                            _Float16* __restrict__ out) {
    const int row = blockIdx.x, tid = threadIdx.x;
    const float* xr = x + (size_t)row * DIM;
    float v0 = xr[tid], v1 = xr[tid + 256];
    __shared__ float red[256];
    red[tid] = v0 + v1;
    __syncthreads();
    for (int off = 128; off > 0; off >>= 1) {
        if (tid < off) red[tid] += red[tid + off];
        __syncthreads();
    }
    float mu = red[0] * (1.0f / DIM);
    __syncthreads();
    float d0 = v0 - mu, d1 = v1 - mu;
    red[tid] = d0 * d0 + d1 * d1;
    __syncthreads();
    for (int off = 128; off > 0; off >>= 1) {
        if (tid < off) red[tid] += red[tid + off];
        __syncthreads();
    }
    float r = rsqrtf(red[0] * (1.0f / DIM) + 1e-5f);
    out[(size_t)row * DIM + tid]       = (_Float16)(d0 * r * g[tid] + b[tid]);
    out[(size_t)row * DIM + tid + 256] = (_Float16)(d1 * r * g[tid + 256] + b[tid + 256]);
}

// ---------------------------------------------------------------- qk l2-norm -> f16
__global__ __launch_bounds__(256) void qknorm_f16_kernel(const float* __restrict__ qk,
                                                         const float* __restrict__ gamma,
                                                         _Float16* __restrict__ out, int nrows) {
    int wave = threadIdx.x >> 5, lane = threadIdx.x & 31;
    int row = blockIdx.x * 8 + wave;
    if (row >= nrows) return;
    const float* p = qk + (size_t)row * DH;
    float a = p[lane], c = p[lane + 32];
    float ss = a * a + c * c;
    #pragma unroll
    for (int off = 16; off > 0; off >>= 1) ss += __shfl_xor(ss, off, 32);
    float r = rsqrtf(ss + 1e-12f) * 8.0f;   // * sqrt(DH)
    out[(size_t)row * DH + lane]      = (_Float16)(a * r * gamma[lane]);
    out[(size_t)row * DH + lane + 32] = (_Float16)(c * r * gamma[lane + 32]);
}

// ---------------------------------------------------------------- row softmax -> f16
__global__ __launch_bounds__(256) void softmax_row_kernel(const float* __restrict__ S,
                                                          _Float16* __restrict__ P, int ncol) {
    const int row = blockIdx.x, tid = threadIdx.x;
    const float* sr = S + (size_t)row * ncol;
    _Float16* pr = P + (size_t)row * ncol;
    __shared__ float red[256];
    float vmax = -1e30f;
    for (int c = tid; c < ncol; c += 256) vmax = fmaxf(vmax, sr[c]);
    red[tid] = vmax;
    __syncthreads();
    for (int off = 128; off > 0; off >>= 1) {
        if (tid < off) red[tid] = fmaxf(red[tid], red[tid + off]);
        __syncthreads();
    }
    float m = red[0];
    __syncthreads();
    float sum = 0.0f;
    for (int c = tid; c < ncol; c += 256) {
        float e = __expf(sr[c] - m);
        sum += e;
        pr[c] = (_Float16)e;
    }
    red[tid] = sum;
    __syncthreads();
    for (int off = 128; off > 0; off >>= 1) {
        if (tid < off) red[tid] += red[tid + off];
        __syncthreads();
    }
    float inv = 1.0f / red[0];
    for (int c = tid; c < ncol; c += 256) pr[c] = (_Float16)((float)pr[c] * inv);
}

// ---------------------------------------------------------------- MoE gate softmax
__global__ __launch_bounds__(256) void gate_kernel(const float* __restrict__ y,
                                                   const float* __restrict__ gW,
                                                   const float* __restrict__ gb,
                                                   float* __restrict__ gate) {
    const int row = blockIdx.x;
    const int e = threadIdx.x >> 5, lane = threadIdx.x & 31;
    const float* yr = y + (size_t)row * DIM;
    float s = 0.0f;
    for (int d = lane; d < DIM; d += 32) s += yr[d] * gW[d * NE + e];
    #pragma unroll
    for (int off = 16; off > 0; off >>= 1) s += __shfl_xor(s, off, 32);
    __shared__ float logits[NE];
    if (lane == 0) logits[e] = s + gb[e];
    __syncthreads();
    if (threadIdx.x == 0) {
        float m = -1e30f;
        for (int i = 0; i < NE; ++i) m = fmaxf(m, logits[i]);
        float ex[NE], sum = 0.0f;
        for (int i = 0; i < NE; ++i) { ex[i] = __expf(logits[i] - m); sum += ex[i]; }
        float inv = 1.0f / sum;
        for (int i = 0; i < NE; ++i) gate[(size_t)row * NE + i] = ex[i] * inv;
    }
}

// ---------------------------------------------------------------- WMMA GEMM
// C[M,N] = A[M,K] @ B[K,N], B supplied COLUMN-MAJOR as Bt[N][K] (ldbt).
// blockIdx.z batches independent GEMMs via element strides aZ/bZ/cZ/chZ.
// Block tile 128 x (32*WNS), 8 waves, wave tile 32 x (16*WNS).
// Staging: GLOBAL_LOAD_ASYNC_TO_LDS_B128 into double-buffered fragment-major LDS
// (Frag[tile16][lane][16 halves]); compute reads are straight ds_load_b128.
// Pipeline: issue tile k+1 -> other buffer, s_wait_asynccnt <batch>, barrier, 2*WNS WMMA.
// mode 0: store C (f32, optional) and Ch (f16 mirror, optional)
// mode 1: v*scale, causal mask (col>row -> -1e30), store C
// mode 2: v = gelu(v + bias[n]); store Ch only
// mode 3: v = (v + bias[n]) * gate[row*gateStride]; C[row,n] += v
template<int WNS>   // 16-col subtiles per wave: 2 (BN=64) or 4 (BN=128)
__global__ __launch_bounds__(256)
void gemm_wmma_kernel(const _Float16* __restrict__ A, int lda, long long aZ,
                      const _Float16* __restrict__ Bt, int ldbt, long long bZ,
                      float* __restrict__ C, long long cZ,
                      _Float16* __restrict__ Ch, long long chZ, int ldc,
                      int M, int N, int K,
                      int mode, float scale,
                      const float* __restrict__ bias,
                      const float* __restrict__ gate, int gateStride) {
    constexpr int BN_ = 32 * WNS;
    __shared__ __attribute__((aligned(32))) _Float16 AsF[2][8][32][16];
    __shared__ __attribute__((aligned(32))) _Float16 BsF[2][2 * WNS][32][16];

    const size_t z = blockIdx.z;
    A += z * aZ;
    Bt += z * bZ;
    if (C)  C += z * cZ;
    if (Ch) Ch += z * chZ;

    const int tid  = threadIdx.x;
    const int wave = tid >> 5;
    const int lane = tid & 31;
    const int wm = wave >> 1;          // 0..3 : 32-row strip
    const int wn = wave & 1;           // 0..1 : (16*WNS)-col strip
    const int blockM0 = blockIdx.y * BM;
    const int blockN0 = blockIdx.x * BN_;
    const int lm = lane & 15;
    const int lg = lane >> 4;

    // --- A staging: 512 chunks of 8 halves, 2 per thread.
    // chunk c of row m covers K[8c..8c+7] -> lane (c&1)*16+(m&15), halves (c>>1)*8
    const int am0 = tid >> 2;          // rows 0..63
    const int am1 = (tid + 256) >> 2;  // rows 64..127
    const int ac8 = tid & 3;
    const _Float16* aP0 = A + (size_t)(blockM0 + am0) * lda + ac8 * 8;
    const _Float16* aP1 = A + (size_t)(blockM0 + am1) * lda + ac8 * 8;
    unsigned ldsA0[2], ldsA1[2];
    #pragma unroll
    for (int bufi = 0; bufi < 2; ++bufi) {
        ldsA0[bufi] = lds_off(&AsF[bufi][am0 >> 4][(ac8 & 1) * 16 + (am0 & 15)][(ac8 >> 1) * 8]);
        ldsA1[bufi] = lds_off(&AsF[bufi][am1 >> 4][(ac8 & 1) * 16 + (am1 & 15)][(ac8 >> 1) * 8]);
    }
    // --- B staging: BN_*4 chunks, WNS/2 per thread.
    // chunk c of col n covers K[8c..8c+7] -> lane (c&1)*16+(n&15), halves (c>>1)*8
    const _Float16* bP[WNS / 2];
    unsigned ldsB[WNS / 2][2];
    #pragma unroll
    for (int j = 0; j < WNS / 2; ++j) {
        int chunk = tid + 256 * j;
        int n = chunk >> 2, c8 = chunk & 3;
        bP[j] = Bt + (size_t)(blockN0 + n) * ldbt + c8 * 8;
        #pragma unroll
        for (int bufi = 0; bufi < 2; ++bufi)
            ldsB[j][bufi] = lds_off(&BsF[bufi][n >> 4][(c8 & 1) * 16 + (n & 15)][(c8 >> 1) * 8]);
    }

    v8f acc[2][WNS];
    #pragma unroll
    for (int sm = 0; sm < 2; ++sm)
        #pragma unroll
        for (int sn = 0; sn < WNS; ++sn) acc[sm][sn] = (v8f){};

    // issue batch for K-tile at offset k into buffer `bufi`
    auto issue = [&](int k, int bufi) {
        async_copy_b128(ldsA0[bufi], aP0 + k);
        async_copy_b128(ldsA1[bufi], aP1 + k);
        #pragma unroll
        for (int j = 0; j < WNS / 2; ++j) async_copy_b128(ldsB[j][bufi], bP[j] + k);
    };

    issue(0, 0);
    for (int k0 = 0; k0 < K; k0 += BK) {
        const int cur = (k0 >> 5) & 1;
        const bool more = (k0 + BK) < K;
        if (more) issue(k0 + BK, cur ^ 1);
        // wait until only the just-issued batch is outstanding -> current tile landed
        if (more) {
            if (WNS == 4) asm volatile("s_wait_asynccnt 0x4" ::: "memory");
            else          asm volatile("s_wait_asynccnt 0x3" ::: "memory");
        } else {
            asm volatile("s_wait_asynccnt 0x0" ::: "memory");
        }
        __syncthreads();

        v16h af[2], bf[WNS];
        af[0] = *(const v16h*)&AsF[cur][2 * wm][lane][0];
        af[1] = *(const v16h*)&AsF[cur][2 * wm + 1][lane][0];
        #pragma unroll
        for (int sn = 0; sn < WNS; ++sn)
            bf[sn] = *(const v16h*)&BsF[cur][WNS * wn + sn][lane][0];
        #pragma unroll
        for (int sm = 0; sm < 2; ++sm)
            #pragma unroll
            for (int sn = 0; sn < WNS; ++sn)
                acc[sm][sn] = __builtin_amdgcn_wmma_f32_16x16x32_f16(
                    false, af[sm], false, bf[sn], (short)0, acc[sm][sn], false, false);
        __syncthreads();
    }

    // epilogue: lane holds col lm, rows lg*8 + r within each 16x16 subtile
    #pragma unroll
    for (int sm = 0; sm < 2; ++sm) {
        #pragma unroll
        for (int sn = 0; sn < WNS; ++sn) {
            const v8f a = acc[sm][sn];
            const int gn = blockN0 + wn * (16 * WNS) + sn * 16 + lm;
            #pragma unroll
            for (int r = 0; r < 8; ++r) {
                const int gm = blockM0 + wm * 32 + sm * 16 + lg * 8 + r;
                float v = a[r];
                if (mode == 0) {
                    if (C)  C[(size_t)gm * ldc + gn] = v;
                    if (Ch) Ch[(size_t)gm * ldc + gn] = (_Float16)v;
                } else if (mode == 1) {
                    v *= scale;
                    if (gn > gm) v = -1e30f;
                    C[(size_t)gm * ldc + gn] = v;
                } else if (mode == 2) {
                    v = gelu_tanh(v + bias[gn]);
                    Ch[(size_t)gm * ldc + gn] = (_Float16)v;
                } else {  // mode 3
                    v = (v + bias[gn]) * gate[(size_t)gm * gateStride];
                    C[(size_t)gm * ldc + gn] += v;
                }
            }
        }
    }
}

// ---------------------------------------------------------------- host orchestration
static inline void* ws_alloc(char* base, size_t& off, size_t bytes) {
    void* p = base + off;
    off += (bytes + 255) & ~(size_t)255;
    return p;
}

extern "C" void kernel_launch(void* const* d_in, const int* in_sizes, int n_in,
                              void* d_out, int out_size, void* d_ws, size_t ws_size,
                              hipStream_t stream) {
    const float* x      = (const float*)d_in[0];
    const float* ln_g   = (const float*)d_in[1];
    const float* ln_b   = (const float*)d_in[2];
    const float* Wq     = (const float*)d_in[3];
    const float* Wk     = (const float*)d_in[4];
    const float* Wv     = (const float*)d_in[5];
    const float* Wo     = (const float*)d_in[6];
    const float* qn_g   = (const float*)d_in[7];
    const float* kn_g   = (const float*)d_in[8];
    const float* gate_W = (const float*)d_in[9];
    const float* gate_b = (const float*)d_in[10];
    const float* W1     = (const float*)d_in[11];
    const float* b1     = (const float*)d_in[12];
    const float* W2     = (const float*)d_in[13];
    const float* b2     = (const float*)d_in[14];
    float* out = (float*)d_out;

    char* base = (char*)d_ws;
    size_t off = 0;
    // NOTE: Wq_t/Wk_t/Wv_t and q_f/k_f/v_f are contiguous equal-size allocations
    // (sizes are multiples of 256B) -> batched QKV GEMM strides over them.
    _Float16* Wq_t  = (_Float16*)ws_alloc(base, off, (size_t)DIM * DIM * 2);   // [N][K]
    _Float16* Wk_t  = (_Float16*)ws_alloc(base, off, (size_t)DIM * DIM * 2);
    _Float16* Wv_t  = (_Float16*)ws_alloc(base, off, (size_t)DIM * DIM * 2);
    _Float16* Wo_t  = (_Float16*)ws_alloc(base, off, (size_t)DIM * DIM * 2);
    _Float16* W1_t  = (_Float16*)ws_alloc(base, off, (size_t)NE * DIM * FF * 2);  // [e][FF][DIM]
    _Float16* W2_t  = (_Float16*)ws_alloc(base, off, (size_t)NE * FF * DIM * 2);  // [e][DIM][FF]
    _Float16* hx_h  = (_Float16*)ws_alloc(base, off, (size_t)SEQ * DIM * 2);
    float*    q_f   = (float*)   ws_alloc(base, off, (size_t)SEQ * DIM * 4);
    float*    k_f   = (float*)   ws_alloc(base, off, (size_t)SEQ * DIM * 4);
    float*    v_f   = (float*)   ws_alloc(base, off, (size_t)SEQ * DIM * 4);
    _Float16* q_h   = (_Float16*)ws_alloc(base, off, (size_t)SEQ * DIM * 2);
    _Float16* k_h   = (_Float16*)ws_alloc(base, off, (size_t)SEQ * DIM * 2);
    _Float16* vT_h  = (_Float16*)ws_alloc(base, off, (size_t)DIM * SEQ * 2);   // [DIM][SEQ]
    float*    S_f   = (float*)   ws_alloc(base, off, (size_t)SEQ * SEQ * 4);
    _Float16* P_all = (_Float16*)ws_alloc(base, off, (size_t)NH * SEQ * SEQ * 2);  // per-head P
    _Float16* o_h   = (_Float16*)ws_alloc(base, off, (size_t)SEQ * DIM * 2);
    float*    y_f   = (float*)   ws_alloc(base, off, (size_t)SEQ * DIM * 4);
    _Float16* y_h   = (_Float16*)ws_alloc(base, off, (size_t)SEQ * DIM * 2);
    float*    gate  = (float*)   ws_alloc(base, off, (size_t)SEQ * NE * 4);
    _Float16* hid_h = (_Float16*)ws_alloc(base, off, (size_t)SEQ * FF * 2);
    (void)ws_size; (void)n_in; (void)in_sizes; (void)out_size;

    // 1. weights -> f16, transposed to column-major [N][K]
    f32_to_f16t_kernel<<<dim3(DIM / 32, DIM / 32, 1), 256, 0, stream>>>(Wq, Wq_t, DIM, DIM);
    f32_to_f16t_kernel<<<dim3(DIM / 32, DIM / 32, 1), 256, 0, stream>>>(Wk, Wk_t, DIM, DIM);
    f32_to_f16t_kernel<<<dim3(DIM / 32, DIM / 32, 1), 256, 0, stream>>>(Wv, Wv_t, DIM, DIM);
    f32_to_f16t_kernel<<<dim3(DIM / 32, DIM / 32, 1), 256, 0, stream>>>(Wo, Wo_t, DIM, DIM);
    f32_to_f16t_kernel<<<dim3(FF / 32, DIM / 32, NE), 256, 0, stream>>>(W1, W1_t, DIM, FF);
    f32_to_f16t_kernel<<<dim3(DIM / 32, FF / 32, NE), 256, 0, stream>>>(W2, W2_t, FF, DIM);

    // 2. layernorm
    layernorm_f16_kernel<<<SEQ, 256, 0, stream>>>(x, ln_g, ln_b, hx_h);

    // 3. QKV projections: single batched launch over {q,k,v}
    gemm_wmma_kernel<4><<<dim3(DIM / 128, SEQ / BM, 3), 256, 0, stream>>>(
        hx_h, DIM, 0, Wq_t, DIM, (long long)DIM * DIM,
        q_f, (long long)SEQ * DIM, nullptr, 0, DIM,
        SEQ, DIM, DIM, 0, 1.0f, nullptr, nullptr, 0);

    // 4. qk-norm (f16 out), v transposed to [DIM][SEQ] f16
    qknorm_f16_kernel<<<(SEQ * NH) / 8, 256, 0, stream>>>(q_f, qn_g, q_h, SEQ * NH);
    qknorm_f16_kernel<<<(SEQ * NH) / 8, 256, 0, stream>>>(k_f, kn_g, k_h, SEQ * NH);
    f32_to_f16t_kernel<<<dim3(DIM / 32, SEQ / 32, 1), 256, 0, stream>>>(v_f, vT_h, SEQ, DIM);

    // 5a. per-head scores + softmax (S_f reused; P stored per head)
    for (int h = 0; h < NH; ++h) {
        gemm_wmma_kernel<4><<<dim3(SEQ / 128, SEQ / BM), 256, 0, stream>>>(
            q_h + h * DH, DIM, 0, k_h + h * DH, DIM, 0, S_f, 0, nullptr, 0, SEQ,
            SEQ, SEQ, DH, 1, 0.125f, nullptr, nullptr, 0);
        softmax_row_kernel<<<SEQ, 256, 0, stream>>>(S_f, P_all + (size_t)h * SEQ * SEQ, SEQ);
    }
    // 5b. o = P @ V for all heads in one batched launch (N=64 -> WNS=2)
    gemm_wmma_kernel<2><<<dim3(1, SEQ / BM, NH), 256, 0, stream>>>(
        P_all, SEQ, (long long)SEQ * SEQ, vT_h, SEQ, (long long)DH * SEQ,
        nullptr, 0, o_h, (long long)DH, DIM,
        SEQ, DH, SEQ, 0, 1.0f, nullptr, nullptr, 0);

    // 6. output projection (f32 for gate + f16 mirror for MoE GEMMs)
    gemm_wmma_kernel<4><<<dim3(DIM / 128, SEQ / BM), 256, 0, stream>>>(
        o_h, DIM, 0, Wo_t, DIM, 0, y_f, 0, y_h, 0, DIM,
        SEQ, DIM, DIM, 0, 1.0f, nullptr, nullptr, 0);

    // 7. MoE gate
    gate_kernel<<<SEQ, 256, 0, stream>>>(y_f, gate_W, gate_b, gate);

    // 8. MoE experts: out = sum_e gate[:,e] * (gelu(y@W1_e + b1_e) @ W2_e + b2_e)
    //    (sequential per-expert accumulate keeps the RMW race-free + deterministic)
    zero_f32_kernel<<<(SEQ * DIM + 255) / 256, 256, 0, stream>>>(out, SEQ * DIM);
    for (int e = 0; e < NE; ++e) {
        gemm_wmma_kernel<4><<<dim3(FF / 128, SEQ / BM), 256, 0, stream>>>(
            y_h, DIM, 0, W1_t + (size_t)e * FF * DIM, DIM, 0,
            nullptr, 0, hid_h, 0, FF,
            SEQ, FF, DIM, 2, 1.0f, b1 + (size_t)e * FF, nullptr, 0);
        gemm_wmma_kernel<4><<<dim3(DIM / 128, SEQ / BM), 256, 0, stream>>>(
            hid_h, FF, 0, W2_t + (size_t)e * DIM * FF, FF, 0,
            out, 0, nullptr, 0, DIM,
            SEQ, DIM, FF, 3, 1.0f, b2 + (size_t)e * DIM, gate + e, NE);
    }
}
